// ArcFace_670014898755
// MI455X (gfx1250) — matrix operand order repaired and checked by reading, not compiled, not run on hardware
//
#include <hip/hip_runtime.h>
#include <hip/hip_bf16.h>
#include <math.h>

typedef __attribute__((ext_vector_type(16))) _Float16 v16h;
typedef __attribute__((ext_vector_type(8)))  float    v8f;
typedef __attribute__((ext_vector_type(4)))  _Float16 v4h;

#define B_ROWS 8192
#define DIM    128
#define N_CLS  16384

// ArcFace constants (margin = 0.5, scale = 64)
__device__ constexpr float kScale = 64.0f;
__device__ constexpr float kCosM  = 0.8775825618903728f;   // cos(0.5)
__device__ constexpr float kSinM  = 0.47942553860420301f;  // sin(0.5)
__device__ constexpr float kTheta = -0.8775825618903728f;  // cos(pi-0.5)
__device__ constexpr float kSinMM = 0.23971276930210151f;  // sin(pi-0.5)*0.5

union FragU { v16h h; uint4 q[2]; };

// A-matrix fragment (16x32 f16, row-major source [rows, 128]).
__device__ inline v16h load_fragA(const _Float16* rowptr, int c, int khalf) {
  FragU f;
  f.q[0] = *(const uint4*)(rowptr + c * 32 + khalf);
  f.q[1] = *(const uint4*)(rowptr + c * 32 + khalf + 16);
  return f.h;
}

// B-matrix fragment (32x16 f16); B = Wn^T so row-major Wn loads contiguously.
__device__ inline v16h load_fragB(const _Float16* rowptr, int c, int k16) {
  FragU f;
  f.q[0] = *(const uint4*)(rowptr + c * 32 + k16);
  f.q[1] = *(const uint4*)(rowptr + c * 32 + k16 + 16);
  return f.h;
}

// Fast acos: A&S 4.4.46 minimax, |err| <= 2e-8 rad; raw v_sqrt_f32.
__device__ inline float fast_acos(float x) {
  const float ax = fabsf(x);
  float p = fmaf(ax, -0.0012624911f, 0.0066700901f);
  p = fmaf(ax, p, -0.0170881256f);
  p = fmaf(ax, p,  0.0308918810f);
  p = fmaf(ax, p, -0.0501743046f);
  p = fmaf(ax, p,  0.0889789874f);
  p = fmaf(ax, p, -0.2145988016f);
  p = fmaf(ax, p,  1.5707963050f);
  const float r = __builtin_amdgcn_sqrtf(1.0f - ax) * p;
  return (x >= 0.0f) ? r : (3.14159265358979f - r);
}

// ---------------- L2 normalize rows, emit f16 ----------------
__global__ void l2norm_f16_kernel(const float* __restrict__ in,
                                  _Float16* __restrict__ out) {
  const int row  = blockIdx.x;
  const int lane = threadIdx.x;  // 0..31
  const float4 v = *(const float4*)(in + (size_t)row * DIM + lane * 4);
  float s = v.x * v.x + v.y * v.y + v.z * v.z + v.w * v.w;
  #pragma unroll
  for (int m = 16; m > 0; m >>= 1) s += __shfl_xor(s, m, 32);
  const float inv = 1.0f / fmaxf(sqrtf(s), 1e-12f);
  v4h h;
  h[0] = (_Float16)(v.x * inv);
  h[1] = (_Float16)(v.y * inv);
  h[2] = (_Float16)(v.z * inv);
  h[3] = (_Float16)(v.w * inv);
  *(v4h*)(out + (size_t)row * DIM + lane * 4) = h;
}

// ---------------- Fused sum-exp GEMM (no label handling) ----------------
// Block = 64 rows x all N columns; 8 waves = 4 row-subtiles x 2 column
// streams (waves of a stream share B tile addresses -> WGP$ reuse).
// Emits per-row sum of exp(64*(clip(cos)-1)) into g_rsum.
__global__ __launch_bounds__(256)
void sumexp_gemm_kernel(const _Float16* __restrict__ en,
                        const _Float16* __restrict__ wn,
                        float* __restrict__ g_rsum) {
  __shared__ float rsum_s[64];
  if (threadIdx.x < 64) rsum_s[threadIdx.x] = 0.0f;
  __syncthreads();

  const int rowblk = blockIdx.x * 64;
  const int wave = threadIdx.x >> 5;
  const int lane = threadIdx.x & 31;
  const int l15  = lane & 15;
  const int hi   = lane >> 4;
  const int rw   = wave & 3;   // row subtile
  const int cs   = wave >> 2;  // column stream
  const int rowbase = rowblk + rw * 16;

  v16h A[4];
  {
    const _Float16* arow = en + (size_t)(rowbase + l15) * DIM;
    #pragma unroll
    for (int c = 0; c < 4; ++c) A[c] = load_fragA(arow, c, hi * 8);
  }

  float sacc[8];
  #pragma unroll
  for (int v = 0; v < 8; ++v) sacc[v] = 0.0f;

  for (int nbase = cs * 16; nbase < N_CLS; nbase += 32) {
    const _Float16* brow = wn + (size_t)(nbase + l15) * DIM;
    v16h Bf[4];
    #pragma unroll
    for (int c = 0; c < 4; ++c) Bf[c] = load_fragB(brow, c, hi * 8);
    if (nbase + 32 < N_CLS)
      __builtin_prefetch((const void*)(wn + (size_t)(nbase + 32) * DIM + lane * 64), 0, 1);

    v8f acc = {};
    #pragma unroll
    for (int c = 0; c < 4; ++c)
      acc = __builtin_amdgcn_wmma_f32_16x16x32_f16(
          false, A[c], false, Bf[c], (short)0, acc, false, false);

    #pragma unroll
    for (int v = 0; v < 8; ++v) {
      const float cv = fminf(fmaxf(acc[v], -1.0f), 1.0f);
      sacc[v] += __expf(kScale * (cv - 1.0f));  // shift-by-64 logsumexp
    }
  }

  #pragma unroll
  for (int v = 0; v < 8; ++v)
    atomicAdd(&rsum_s[rw * 16 + v + hi * 8], sacc[v]);
  __syncthreads();
  if (threadIdx.x < 64)
    atomicAdd(&g_rsum[rowblk + threadIdx.x], rsum_s[threadIdx.x]);
}

// ---------------- CE fixup: margin on target class, logsumexp, mean -------
// One wave per row: recompute target cosine (f32 dot of f16 inputs),
// swap exp(cos_t) -> exp(final) in the row sum, emit CE contribution.
__global__ __launch_bounds__(256)
void ce_fixup_kernel(const _Float16* __restrict__ en,
                     const _Float16* __restrict__ wn,
                     const int* __restrict__ labels,
                     const float* __restrict__ g_rsum,
                     float* __restrict__ accum) {
  const int row  = blockIdx.x * 8 + (threadIdx.x >> 5);
  const int lane = threadIdx.x & 31;
  const _Float16* erow = en + (size_t)row * DIM;
  const _Float16* wrow = wn + (size_t)labels[row] * DIM;
  const v4h e = *(const v4h*)(erow + lane * 4);
  const v4h w = *(const v4h*)(wrow + lane * 4);
  float s = (float)e[0] * (float)w[0] + (float)e[1] * (float)w[1] +
            (float)e[2] * (float)w[2] + (float)e[3] * (float)w[3];
  #pragma unroll
  for (int m = 16; m > 0; m >>= 1) s += __shfl_xor(s, m, 32);
  if (lane == 0) {
    const float cos_t  = fminf(fmaxf(s, -1.0f), 1.0f);
    const float s_t    = sqrtf(fmaxf(1.0f - cos_t * cos_t, 0.0f));
    const float cos_tm = cos_t * kCosM - s_t * kSinM;
    const float fin    = (cos_t > kTheta) ? cos_tm : (cos_t - kSinMM);
    const float sum    = g_rsum[row] - __expf(kScale * (cos_t - 1.0f))
                                     + __expf(kScale * (fin   - 1.0f));
    const float ce = kScale + __logf(sum) - kScale * fin;
    atomicAdd(&accum[0], ce);
  }
}

// ---------------- Fused pairwise-angle GEMM: sum_{a<b} acos(wn_a.wn_b) ----
__global__ __launch_bounds__(256)
void pair_acos_kernel(const _Float16* __restrict__ wn,
                      float* __restrict__ accum) {
  __shared__ float bsum_s;
  if (threadIdx.x == 0) bsum_s = 0.0f;
  __syncthreads();

  const int rowblk = blockIdx.x * 64;
  const int wave = threadIdx.x >> 5;
  const int lane = threadIdx.x & 31;
  const int l15  = lane & 15;
  const int hi   = lane >> 4;
  const int rw   = wave & 3;
  const int cs   = wave >> 2;
  const int rowbase = rowblk + rw * 16;

  v16h A[4];
  {
    const _Float16* arow = wn + (size_t)(rowbase + l15) * DIM;
    #pragma unroll
    for (int c = 0; c < 4; ++c) A[c] = load_fragA(arow, c, hi * 8);
  }

  float psum = 0.0f;
  const int nb0 = (rowblk / 32) * 32;  // skip chunks fully below diagonal

  for (int nbase = nb0 + cs * 16; nbase < N_CLS; nbase += 32) {
    const _Float16* brow = wn + (size_t)(nbase + l15) * DIM;
    v16h Bf[4];
    #pragma unroll
    for (int c = 0; c < 4; ++c) Bf[c] = load_fragB(brow, c, hi * 8);
    if (nbase + 32 < N_CLS)
      __builtin_prefetch((const void*)(wn + (size_t)(nbase + 32) * DIM + lane * 64), 0, 1);

    v8f acc = {};
    #pragma unroll
    for (int c = 0; c < 4; ++c)
      acc = __builtin_amdgcn_wmma_f32_16x16x32_f16(
          false, A[c], false, Bf[c], (short)0, acc, false, false);

    const int col = nbase + l15;
    #pragma unroll
    for (int v = 0; v < 8; ++v) {
      const int row = rowbase + v + hi * 8;
      const float cv = fminf(fmaxf(acc[v], -1.0f + 1e-6f), 1.0f - 1e-6f);
      const float a  = fast_acos(cv);
      psum += (col > row) ? a : 0.0f;  // branchless upper-triangle mask
    }
  }

  atomicAdd(&bsum_s, psum);
  __syncthreads();
  if (threadIdx.x == 0) atomicAdd(&accum[1], bsum_s);
}

// ---------------- tiny helpers ----------------
__global__ void zero_ws_kernel(float* accum, float* g_rsum) {
  const int idx = blockIdx.x * 256 + threadIdx.x;
  if (idx < 2) accum[idx] = 0.0f;
  if (idx < B_ROWS) g_rsum[idx] = 0.0f;
}

__global__ void finalize_kernel(const float* __restrict__ accum,
                                float* __restrict__ out) {
  const float ce = accum[0] / (float)B_ROWS;
  const float l6 = accum[1] / (-3.14f * (float)(N_CLS - 1));
  out[0] = ce + 10.0f * l6;
}

extern "C" void kernel_launch(void* const* d_in, const int* in_sizes, int n_in,
                              void* d_out, int out_size, void* d_ws, size_t ws_size,
                              hipStream_t stream) {
  const float* emb    = (const float*)d_in[0];  // [B, D] f32
  const int*   labels = (const int*)  d_in[1];  // [B] i32
  const float* weight = (const float*)d_in[2];  // [N, D] f32

  char* ws = (char*)d_ws;
  float*    accum  = (float*)ws;                                   // 2 floats
  float*    g_rsum = (float*)(ws + 256);                           // 32 KB
  _Float16* en     = (_Float16*)(ws + 256 + B_ROWS * 4);           // 2 MB
  _Float16* wnn    = (_Float16*)(ws + 256 + B_ROWS * 4 +
                                 (size_t)B_ROWS * DIM * 2);        // 4 MB

  zero_ws_kernel<<<B_ROWS / 256, 256, 0, stream>>>(accum, g_rsum);
  l2norm_f16_kernel<<<B_ROWS, 32, 0, stream>>>(emb, en);
  l2norm_f16_kernel<<<N_CLS, 32, 0, stream>>>(weight, wnn);
  sumexp_gemm_kernel<<<B_ROWS / 64, 256, 0, stream>>>(en, wnn, g_rsum);
  ce_fixup_kernel<<<B_ROWS / 8, 256, 0, stream>>>(en, wnn, labels, g_rsum, accum);
  pair_acos_kernel<<<N_CLS / 64, 256, 0, stream>>>(wnn, accum);
  finalize_kernel<<<1, 1, 0, stream>>>(accum, (float*)d_out);
}
